// CustomEncoderLayer_14181982011891
// MI455X (gfx1250) — matrix operand order repaired and checked by reading, not compiled, not run on hardware
//
#include <hip/hip_runtime.h>
#include <hip/hip_bf16.h>
#include <math.h>
#include <stdint.h>

typedef __attribute__((ext_vector_type(16))) _Float16 v16h;
typedef __attribute__((ext_vector_type(8)))  _Float16 v8h;
typedef __attribute__((ext_vector_type(8)))  float    v8f;

#define B_  4
#define T_  2048
#define D_  1024
#define NH_ 16
#define DK_ 64
#define DF_ 4096
#define M_  (B_*T_)   // 8192 tokens

static __device__ __forceinline__ v8f wmma_f16(v16h a, v16h b, v8f c) {
  // v_wmma_f32_16x16x32_f16  D = A x B + C
  return __builtin_amdgcn_wmma_f32_16x16x32_f16(false, a, false, b, (short)0, c, false, false);
}

// ---------------------------------------------------------------- fp32 -> f16
__global__ void cvt_f32_f16(const float* __restrict__ in, _Float16* __restrict__ out, int n) {
  int i = blockIdx.x * blockDim.x + threadIdx.x;
  if (i < n) out[i] = (_Float16)in[i];
}

// ---------------------------------------------------------------- LayerNorm (row = 1024)
__global__ __launch_bounds__(256) void layernorm_f16(const float* __restrict__ x,
                                                     const float* __restrict__ g,
                                                     const float* __restrict__ be,
                                                     _Float16* __restrict__ out) {
  __shared__ float red[256];
  int row = blockIdx.x;
  const float* xr = x + (size_t)row * D_;
  float v0[4];
  float s = 0.f;
#pragma unroll
  for (int j = 0; j < 4; ++j) { v0[j] = xr[threadIdx.x + j*256]; s += v0[j]; }
  red[threadIdx.x] = s; __syncthreads();
  for (int off = 128; off > 0; off >>= 1) {
    if (threadIdx.x < off) red[threadIdx.x] += red[threadIdx.x + off];
    __syncthreads();
  }
  float mean = red[0] * (1.0f / D_);
  __syncthreads();
  float s2 = 0.f;
#pragma unroll
  for (int j = 0; j < 4; ++j) { float d = v0[j] - mean; s2 += d * d; }
  red[threadIdx.x] = s2; __syncthreads();
  for (int off = 128; off > 0; off >>= 1) {
    if (threadIdx.x < off) red[threadIdx.x] += red[threadIdx.x + off];
    __syncthreads();
  }
  float rstd = rsqrtf(red[0] * (1.0f / D_) + 1e-5f);
#pragma unroll
  for (int j = 0; j < 4; ++j) {
    int c = threadIdx.x + j*256;
    out[(size_t)row * D_ + c] = (_Float16)((v0[j] - mean) * rstd * g[c] + be[c]);
  }
}

// ---------------------------------------------------------------- RoPE (in-place, [BH][T][64] f16)
__global__ void rope_kernel(_Float16* __restrict__ buf) {
  int idx = blockIdx.x * blockDim.x + threadIdx.x;  // B*NH*T*32 pair-threads
  int i  = idx & 31;
  int t  = (idx >> 5) & (T_ - 1);
  int bh = idx >> 16;
  if (bh >= B_ * NH_) return;
  float theta = __powf(10000.0f, -(2.0f * (float)i) / 64.0f);
  float ang = (float)t * theta;
  float sn, cs; __sincosf(ang, &sn, &cs);
  _Float16* p = buf + ((size_t)bh * T_ + t) * DK_ + 2 * i;
  float x1 = (float)p[0], x2 = (float)p[1];
  p[0] = (_Float16)(x1 * cs - x2 * sn);
  p[1] = (_Float16)(x1 * sn + x2 * cs);
}

// ---------------------------------------------------------------- WMMA GEMM: out = A(MxK) * W(NoutxK)^T + bias
// block = 256 thr (8 waves as 4 row-groups x 2 col-groups); block tile 128x128.
// wave tile 32x64: 2 A-frags x 4 B-frags -> 8 WMMA per K-step (1.0 loads/WMMA).
constexpr int EPI_QKV   = 0;  // f16 out, scatter to [B][H][T][64]
constexpr int EPI_RES32 = 1;  // fp32 out = resid + acc
constexpr int EPI_GELU  = 2;  // f16 out = gelu(acc)

template <int EPI>
__global__ __launch_bounds__(256) void gemm_wmma(const _Float16* __restrict__ A,
                                                 const _Float16* __restrict__ W,
                                                 const float* __restrict__ bias,
                                                 const float* __restrict__ resid,
                                                 void* __restrict__ outp,
                                                 int M, int K, int Nout) {
  int lane = threadIdx.x & 31;
  int wv   = threadIdx.x >> 5;
  int mBase = blockIdx.y * 128 + (wv >> 1) * 32;   // two 16-row tiles
  int nBase = blockIdx.x * 128 + (wv & 1) * 64;    // four 16-col tiles

  // A fragment: lane<16 -> M=lane, K {0..15}; lane>=16 -> M=lane-16, K {16..31}
  const _Float16* arow0 = A + (size_t)(mBase + (lane & 15)) * K + ((lane & 16) ? 16 : 0);
  const _Float16* arow1 = arow0 + (size_t)16 * K;
  // B fragment: lane column n, 16 consecutive K values at (lane&16?16:0) -> contiguous in W row n
  int kb2 = (lane & 16) ? 16 : 0;
  const _Float16* w0 = W + (size_t)(nBase +  0 + (lane & 15)) * K + kb2;
  const _Float16* w1 = W + (size_t)(nBase + 16 + (lane & 15)) * K + kb2;
  const _Float16* w2 = W + (size_t)(nBase + 32 + (lane & 15)) * K + kb2;
  const _Float16* w3 = W + (size_t)(nBase + 48 + (lane & 15)) * K + kb2;

  v8f acc[2][4] = {};
  for (int k0 = 0; k0 < K; k0 += 32) {
    v8h lo0 = *(const v8h*)(arow0 + k0);
    v8h hi0 = *(const v8h*)(arow0 + k0 + 8);
    v8h lo1 = *(const v8h*)(arow1 + k0);
    v8h hi1 = *(const v8h*)(arow1 + k0 + 8);
    v16h a0, a1;
#pragma unroll
    for (int j = 0; j < 8; ++j) {
      a0[j] = lo0[j]; a0[j + 8] = hi0[j];
      a1[j] = lo1[j]; a1[j + 8] = hi1[j];
    }
    v16h b0 = *(const v16h*)(w0 + k0);
    v16h b1 = *(const v16h*)(w1 + k0);
    v16h b2 = *(const v16h*)(w2 + k0);
    v16h b3 = *(const v16h*)(w3 + k0);
    acc[0][0] = wmma_f16(a0, b0, acc[0][0]);
    acc[0][1] = wmma_f16(a0, b1, acc[0][1]);
    acc[0][2] = wmma_f16(a0, b2, acc[0][2]);
    acc[0][3] = wmma_f16(a0, b3, acc[0][3]);
    acc[1][0] = wmma_f16(a1, b0, acc[1][0]);
    acc[1][1] = wmma_f16(a1, b1, acc[1][1]);
    acc[1][2] = wmma_f16(a1, b2, acc[1][2]);
    acc[1][3] = wmma_f16(a1, b3, acc[1][3]);
  }

  int rowHalf = (lane >> 4) * 8;   // C/D layout: VGPR r holds M = r + 8*(lane>>4), N = lane&15
#pragma unroll
  for (int rt = 0; rt < 2; ++rt) {
#pragma unroll
    for (int nt = 0; nt < 4; ++nt) {
      int n = nBase + nt * 16 + (lane & 15);
      float bn = bias ? bias[n] : 0.0f;
#pragma unroll
      for (int r = 0; r < 8; ++r) {
        int m = mBase + rt * 16 + rowHalf + r;
        float v = acc[rt][nt][r] + bn;
        if (EPI == EPI_QKV) {
          int bb = m >> 11, t = m & (T_ - 1);
          int head = n >> 6, d = n & 63;
          ((_Float16*)outp)[(((size_t)bb * NH_ + head) * T_ + t) * DK_ + d] = (_Float16)v;
        } else if (EPI == EPI_RES32) {
          ((float*)outp)[(size_t)m * Nout + n] = resid[(size_t)m * Nout + n] + v;
        } else { // EPI_GELU
          float gg = 0.5f * v * (1.0f + erff(v * 0.70710678f));
          ((_Float16*)outp)[(size_t)m * Nout + n] = (_Float16)gg;
        }
      }
    }
  }
}

// ---------------------------------------------------------------- Flash attention
// grid (T/64, B*NH), block 128 (4 waves); wave = 16 q rows; kv chunks of 32.
// K tile staged with CDNA5 async copy (global -> LDS, ASYNCcnt); V staged transposed via VGPRs.
__global__ __launch_bounds__(128) void flash_attn(const _Float16* __restrict__ q,
                                                  const _Float16* __restrict__ k,
                                                  const _Float16* __restrict__ v,
                                                  const unsigned char* __restrict__ mask,
                                                  _Float16* __restrict__ out) {
  __shared__ __align__(32) _Float16 Ks[32][64];      // K chunk, row major [key][d]
  __shared__ __align__(32) _Float16 Vt[64][32];      // V chunk, transposed [d][key]
  __shared__ __align__(32) _Float16 Ps[4][16][32];   // per-wave P tile

  int lane = threadIdx.x & 31;
  int wv   = threadIdx.x >> 5;
  int bh = blockIdx.y;
  int bb = bh >> 4;         // NH_ = 16
  int hh = bh & 15;
  size_t seqBase = (size_t)bh * T_;

  // load q fragments (persist over kv loop): rows qBase..qBase+15, d = 0..63 (two K-steps)
  int qRow = blockIdx.x * 64 + wv * 16 + (lane & 15);
  const _Float16* qr = q + (seqBase + qRow) * DK_ + ((lane & 16) ? 16 : 0);
  v8h q0 = *(const v8h*)(qr);
  v8h q1 = *(const v8h*)(qr + 8);
  v8h q2 = *(const v8h*)(qr + 32);
  v8h q3 = *(const v8h*)(qr + 40);
  v16h aq0, aq1;
#pragma unroll
  for (int j = 0; j < 8; ++j) {
    aq0[j] = q0[j]; aq0[j + 8] = q1[j];
    aq1[j] = q2[j]; aq1[j + 8] = q3[j];
  }

  float m_i[8], l_i[8];
#pragma unroll
  for (int r = 0; r < 8; ++r) { m_i[r] = -1e30f; l_i[r] = 0.0f; }
  v8f accd[4] = {};

  const unsigned char* mrow = mask + (size_t)bb * T_;

  for (int kv0 = 0; kv0 < T_; kv0 += 32) {
    // stage: Ks via async global->LDS B128 copies (no VGPR round trip), Vt transposed via VGPRs.
#pragma unroll
    for (int c = 0; c < 2; ++c) {
      int cc  = threadIdx.x * 2 + c;     // 0..255
      int row = cc >> 3;                 // key 0..31
      int col = (cc & 7) * 8;            // d offset
      const _Float16* gk = k + (seqBase + kv0 + row) * DK_ + col;
      unsigned ldsK = (unsigned)(uintptr_t)&Ks[row][col];  // generic->LDS = low 32 bits
      asm volatile("global_load_async_to_lds_b128 %0, %1, off"
                   :: "v"(ldsK), "v"(gk) : "memory");
      v8h vv8 = *(const v8h*)(v + (seqBase + kv0 + row) * DK_ + col);
#pragma unroll
      for (int j = 0; j < 8; ++j) Vt[col + j][row] = vv8[j];
    }
    asm volatile("s_wait_asynccnt 0x0" ::: "memory");
    __syncthreads();

    // scores 16x32 = q(16x64) . K^T  (two WMMA K-steps over d, two n-tiles)
    int key = lane & 15;
    int dkb = (lane & 16) ? 16 : 0;
    float p[2][8];
#pragma unroll
    for (int nt = 0; nt < 2; ++nt) {
      v16h bk0 = *(const v16h*)&Ks[nt * 16 + key][dkb];
      v16h bk1 = *(const v16h*)&Ks[nt * 16 + key][32 + dkb];
      v8f sc = {};
      sc = wmma_f16(aq0, bk0, sc);
      sc = wmma_f16(aq1, bk1, sc);
      bool masked = (mrow[kv0 + nt * 16 + key] != 0);
#pragma unroll
      for (int r = 0; r < 8; ++r) p[nt][r] = masked ? -1e30f : sc[r] * 0.125f;
    }

    // online softmax (row lives across the 16 lanes of each half-wave; xor masks < 16)
#pragma unroll
    for (int r = 0; r < 8; ++r) {
      float tm = fmaxf(p[0][r], p[1][r]);
#pragma unroll
      for (int off = 8; off >= 1; off >>= 1) tm = fmaxf(tm, __shfl_xor(tm, off, 32));
      float newm = fmaxf(m_i[r], tm);
      float e0 = __expf(p[0][r] - newm);
      float e1 = __expf(p[1][r] - newm);
      float corr = __expf(m_i[r] - newm);
      m_i[r] = newm;
      p[0][r] = e0; p[1][r] = e1;
      float rs = e0 + e1;
#pragma unroll
      for (int off = 8; off >= 1; off >>= 1) rs += __shfl_xor(rs, off, 32);
      l_i[r] = l_i[r] * corr + rs;
#pragma unroll
      for (int dt = 0; dt < 4; ++dt) accd[dt][r] *= corr;
    }

    // bounce P (C layout) -> LDS -> A layout, then P(16x32) . V(32x64)
    int prow = (lane >> 4) * 8;
#pragma unroll
    for (int nt = 0; nt < 2; ++nt)
#pragma unroll
      for (int r = 0; r < 8; ++r)
        Ps[wv][prow + r][nt * 16 + key] = (_Float16)p[nt][r];
    asm volatile("s_wait_dscnt 0x0" ::: "memory");

    int pm = lane & 15;
    int pb = (lane & 16) ? 16 : 0;
    v8h pl0 = *(const v8h*)&Ps[wv][pm][pb];
    v8h pl1 = *(const v8h*)&Ps[wv][pm][pb + 8];
    v16h aP;
#pragma unroll
    for (int j = 0; j < 8; ++j) { aP[j] = pl0[j]; aP[j + 8] = pl1[j]; }

    int nb2 = (lane & 16) ? 16 : 0;
#pragma unroll
    for (int dt = 0; dt < 4; ++dt) {
      v16h bv = *(const v16h*)&Vt[dt * 16 + (lane & 15)][nb2];
      accd[dt] = wmma_f16(aP, bv, accd[dt]);
    }
    __syncthreads();
  }

  // normalize + write [B][T][D] head-concatenated (f16)
  int rowHalf = (lane >> 4) * 8;
#pragma unroll
  for (int dt = 0; dt < 4; ++dt)
#pragma unroll
    for (int r = 0; r < 8; ++r) {
      int t = blockIdx.x * 64 + wv * 16 + rowHalf + r;
      float o = accd[dt][r] / l_i[r];
      out[((size_t)(bb * T_ + t)) * D_ + hh * DK_ + dt * 16 + (lane & 15)] = (_Float16)o;
    }
}

// ---------------------------------------------------------------- launch
extern "C" void kernel_launch(void* const* d_in, const int* in_sizes, int n_in,
                              void* d_out, int out_size, void* d_ws, size_t ws_size,
                              hipStream_t stream) {
  (void)in_sizes; (void)n_in; (void)out_size; (void)ws_size;
  const float* x    = (const float*)d_in[0];
  const unsigned char* pmask = (const unsigned char*)d_in[1];
  const float* Wq = (const float*)d_in[2];
  const float* bq = (const float*)d_in[3];
  const float* Wk = (const float*)d_in[4];
  const float* bk = (const float*)d_in[5];
  const float* Wv = (const float*)d_in[6];
  const float* bv = (const float*)d_in[7];
  const float* Wo = (const float*)d_in[8];
  const float* bo = (const float*)d_in[9];
  const float* W1 = (const float*)d_in[10];
  const float* b1 = (const float*)d_in[11];
  const float* W2 = (const float*)d_in[12];
  const float* b2 = (const float*)d_in[13];
  const float* g1 = (const float*)d_in[14];
  const float* be1= (const float*)d_in[15];
  const float* g2 = (const float*)d_in[16];
  const float* be2= (const float*)d_in[17];
  float* out = (float*)d_out;

  // workspace layout (peak ~152 MiB, overlaid by stream order)
  char* ws = (char*)d_ws;
  const size_t MB = 1024ull * 1024ull;
  _Float16* Wq16 = (_Float16*)(ws + 0 * MB);
  _Float16* Wk16 = (_Float16*)(ws + 2 * MB);
  _Float16* Wv16 = (_Float16*)(ws + 4 * MB);
  _Float16* Wo16 = (_Float16*)(ws + 6 * MB);
  _Float16* W116 = (_Float16*)(ws + 8 * MB);
  _Float16* W216 = (_Float16*)(ws + 16 * MB);
  _Float16* hf16 = (_Float16*)(ws + 24 * MB);   // 16MB: h -> attn -> h2 (sequential reuse)
  _Float16* q16  = (_Float16*)(ws + 40 * MB);   // 16MB
  _Float16* k16  = (_Float16*)(ws + 56 * MB);   // 16MB
  _Float16* v16  = (_Float16*)(ws + 72 * MB);   // 16MB
  float*    x2   = (float*)   (ws + 40 * MB);   // 32MB, reuses q/k after attention
  _Float16* ff16 = (_Float16*)(ws + 88 * MB);   // 64MB
  _Float16* attn16 = hf16;
  _Float16* h2f16  = hf16;

  const int thr = 256;
  cvt_f32_f16<<<(D_*D_ + thr-1)/thr, thr, 0, stream>>>(Wq, Wq16, D_*D_);
  cvt_f32_f16<<<(D_*D_ + thr-1)/thr, thr, 0, stream>>>(Wk, Wk16, D_*D_);
  cvt_f32_f16<<<(D_*D_ + thr-1)/thr, thr, 0, stream>>>(Wv, Wv16, D_*D_);
  cvt_f32_f16<<<(D_*D_ + thr-1)/thr, thr, 0, stream>>>(Wo, Wo16, D_*D_);
  cvt_f32_f16<<<(DF_*D_ + thr-1)/thr, thr, 0, stream>>>(W1, W116, DF_*D_);
  cvt_f32_f16<<<(D_*DF_ + thr-1)/thr, thr, 0, stream>>>(W2, W216, D_*DF_);

  layernorm_f16<<<M_, 256, 0, stream>>>(x, g1, be1, hf16);

  dim3 gq(D_ / 128, M_ / 128);                 // 128x128 block tiles
  gemm_wmma<EPI_QKV><<<gq, 256, 0, stream>>>(hf16, Wq16, bq, nullptr, q16, M_, D_, D_);
  gemm_wmma<EPI_QKV><<<gq, 256, 0, stream>>>(hf16, Wk16, bk, nullptr, k16, M_, D_, D_);
  gemm_wmma<EPI_QKV><<<gq, 256, 0, stream>>>(hf16, Wv16, bv, nullptr, v16, M_, D_, D_);

  const int nrope = B_ * NH_ * T_ * 32;
  rope_kernel<<<nrope / 256, 256, 0, stream>>>(q16);
  rope_kernel<<<nrope / 256, 256, 0, stream>>>(k16);

  dim3 gf(T_ / 64, B_ * NH_);
  flash_attn<<<gf, 128, 0, stream>>>(q16, k16, v16, pmask, attn16);

  gemm_wmma<EPI_RES32><<<gq, 256, 0, stream>>>(attn16, Wo16, bo, x, x2, M_, D_, D_);

  layernorm_f16<<<M_, 256, 0, stream>>>(x2, g2, be2, h2f16);

  dim3 g1d(DF_ / 128, M_ / 128);
  gemm_wmma<EPI_GELU><<<g1d, 256, 0, stream>>>(h2f16, W116, b1, nullptr, ff16, M_, D_, DF_);

  gemm_wmma<EPI_RES32><<<gq, 256, 0, stream>>>(ff16, W216, b2, x2, out, M_, DF_, D_);
}
